// latent_SDE_72301479460974
// MI455X (gfx1250) — matrix operand order repaired and checked by reading, not compiled, not run on hardware
//
#include <hip/hip_runtime.h>
#include <math.h>

// ---------------------------------------------------------------------------
// Latent SDE (Euler-Maruyama, 200 steps) on gfx1250 via v_wmma_f32_16x16x32_bf16
//   z in f32 accumulator registers for all 200 steps (16 batch rows per wave)
//   GEMM inputs rounded to bf16; weights pre-transposed to bf16 fragments.
//   einsum("bdk,bk->bd", g, dW) fused as K=256 GEMM with A2[b,2c+k]=z[b,c]*dW[b,k].
//   Phases run kb-outer / tile-inner: each LDS A-fragment loaded once per phase,
//   reused across all 8 B tiles (8x less LDS traffic, bounded register pressure).
//   GELU: branchless vectorized erf (v_pk_fma chains; no EXEC divergence).
//   Biases stay in LDS: per-step ds_load_2addr reloads are cheaper than the
//   register pressure of pinning them.
// ---------------------------------------------------------------------------

typedef __bf16 bf16_t;
typedef __bf16 v16bf __attribute__((ext_vector_type(16)));
typedef __bf16 v8bf  __attribute__((ext_vector_type(8)));
typedef float  v8f   __attribute__((ext_vector_type(8)));

#define BATCH     32768
#define NSTEPS    200
#define DT_F      0.005f
#define SQRT_DT_F 0.07071067811865475f

#define WAVES_PER_BLOCK 4
#define ROWS_PER_BLOCK  (WAVES_PER_BLOCK * 16)

// d_ws bf16 layout (element offsets)
#define OFF_ENCT 0        // 128x128 encT[n][c]
#define OFF_MU1T 16384    // 128x128 mu1T[n][c]
#define OFF_MU2T 32768    // 128x128 mu2T[n][c]
#define OFF_DECT 49152    // 128x128 decT[n][c]
#define OFF_W2T  65536    // 128x256 w2T[d][2c+k] = sig_w[c][2d+k]
#define WTS_ELEMS 98304

__device__ __forceinline__ v8f wmma_bf16(v16bf a, v16bf b, v8f c) {
  return __builtin_amdgcn_wmma_f32_16x16x32_bf16(
      /*neg_a=*/false, a, /*neg_b=*/false, b,
      /*c_mod=*/(short)0, c, /*reuse_a=*/false, /*reuse_b=*/false);
}

// A fragment (16x32) from an LDS row-major tile [16][ldk] bf16, K-block kb.
// Layout: lanes 0-15 hold K 0..15, lanes 16-31 hold K 16..31 (matches B fragment).
__device__ __forceinline__ v16bf lds_a_frag(const bf16_t* base, int ldk, int kb, int lane) {
  const int row = lane & 15;
  const int off = (lane & 16) ? 16 : 0;
  const bf16_t* p0 = base + row * ldk + kb * 32 + off;
  union { v16bf v; v8bf h[2]; } u;
  u.h[0] = *(const v8bf*)p0;
  u.h[1] = *(const v8bf*)(p0 + 8);
  return u.v;
}

// B fragment (32x16) from global transposed weight wT (rows of ldk bf16).
// Lane holds 16 consecutive K of one column n; lanes 0-15 K=0..15, 16-31 K=16..31.
__device__ __forceinline__ v16bf glb_b_frag(const bf16_t* __restrict__ wT, int ldk,
                                            int t, int kb, int lane) {
  const int n = t * 16 + (lane & 15);
  const int kbase = kb * 32 + ((lane & 16) ? 16 : 0);
  return *(const v16bf*)(wT + (size_t)n * ldk + kbase);
}

// One GEMM phase over 8 output tiles: kb-outer / tile-inner so each A fragment
// is loaded from LDS exactly once and reused for all 8 WMMAs.
template <int KB>
__device__ __forceinline__ void gemm_tiles(const bf16_t* aBase, int aLdk,
                                           const bf16_t* __restrict__ wT, int wLdk,
                                           int lane, v8f acc[8]) {
#pragma unroll
  for (int kb = 0; kb < KB; ++kb) {
    v16bf a = lds_a_frag(aBase, aLdk, kb, lane);
#pragma unroll
    for (int t = 0; t < 8; ++t)
      acc[t] = wmma_bf16(a, glb_b_frag(wT, wLdk, t, kb, lane), acc[t]);
  }
}

// Exact-GELU on a whole accumulator (branchless A&S 7.1.26 erf, |err|<=1.5e-7).
// Vector arithmetic -> v_pk_fma_f32 chains; only rcp/exp are per-element trans.
__device__ __forceinline__ v8f gelu8(v8f x) {
  v8f xs = x * 0.7071067811865476f;                 // x / sqrt(2)
  v8f ax;
#pragma unroll
  for (int i = 0; i < 8; ++i) ax[i] = __builtin_fabsf(xs[i]);
  v8f t;
#pragma unroll
  for (int i = 0; i < 8; ++i)
    t[i] = __builtin_amdgcn_rcpf(__builtin_fmaf(0.3275911f, ax[i], 1.0f));
  v8f p = t * 1.061405429f + (-1.453152027f);
  p = p * t + 1.421413741f;
  p = p * t + (-0.284496736f);
  p = p * t + 0.254829592f;
  p = p * t;
  v8f e;
#pragma unroll
  for (int i = 0; i < 8; ++i)
    e[i] = __builtin_amdgcn_exp2f(ax[i] * ax[i] * -1.4426950408889634f);
  v8f er = 1.0f - p * e;                            // erf(|xs|)
#pragma unroll
  for (int i = 0; i < 8; ++i) er[i] = __builtin_copysignf(er[i], xs[i]);
  return 0.5f * x * (1.0f + er);
}

// -------------------- weight prep: f32 -> bf16, transposed --------------------
__global__ void sde_prep_kernel(const float* __restrict__ enc_w,
                                const float* __restrict__ mu_w1,
                                const float* __restrict__ mu_w2,
                                const float* __restrict__ dec_w,
                                const float* __restrict__ sig_w,
                                bf16_t* __restrict__ wts) {
  int idx = blockIdx.x * blockDim.x + threadIdx.x;
  if (idx < OFF_MU1T) {                    // encT[n][c] = enc_w[c][n]
    int j = idx - OFF_ENCT; int n = j >> 7, c = j & 127;
    wts[idx] = (bf16_t)enc_w[c * 128 + n];
  } else if (idx < OFF_MU2T) {
    int j = idx - OFF_MU1T; int n = j >> 7, c = j & 127;
    wts[idx] = (bf16_t)mu_w1[c * 128 + n];
  } else if (idx < OFF_DECT) {
    int j = idx - OFF_MU2T; int n = j >> 7, c = j & 127;
    wts[idx] = (bf16_t)mu_w2[c * 128 + n];
  } else if (idx < OFF_W2T) {
    int j = idx - OFF_DECT; int n = j >> 7, c = j & 127;
    wts[idx] = (bf16_t)dec_w[c * 128 + n];
  } else if (idx < WTS_ELEMS) {            // w2T[d][2c+k] = sig_w[c][2d+k]
    int j = idx - OFF_W2T; int d = j >> 8; int kk = j & 255;
    int c = kk >> 1, k = kk & 1;
    wts[idx] = (bf16_t)sig_w[c * 256 + 2 * d + k];
  }
}

// -------------------- main persistent SDE kernel --------------------
__global__ __launch_bounds__(128) __attribute__((amdgpu_waves_per_eu(2)))
void sde_kernel(
    const float* __restrict__ y, const float* __restrict__ noise,
    const float* __restrict__ enc_b, const float* __restrict__ mu_b1,
    const float* __restrict__ mu_b2, const float* __restrict__ sig_b,
    const float* __restrict__ dec_b, const bf16_t* __restrict__ wts,
    float* __restrict__ out) {
  // Wave-private LDS staging (no inter-wave sharing -> no loop barriers).
  __shared__ __align__(16) bf16_t s_zbf[WAVES_PER_BLOCK][16 * 128]; // old z, bf16
  __shared__ __align__(16) bf16_t s_a2 [WAVES_PER_BLOCK][16 * 256]; // z*dW pairs; h overlays
  __shared__ float s_dw[WAVES_PER_BLOCK][32];
  __shared__ float s_b1[128], s_b2[128], s_encb[128], s_decb[128], s_sigb[256];

  const int tid  = threadIdx.x;
  const int wave = tid >> 5;
  const int lane = tid & 31;

  s_b1[tid]   = mu_b1[tid];
  s_b2[tid]   = mu_b2[tid];
  s_encb[tid] = enc_b[tid];
  s_decb[tid] = dec_b[tid];
  s_sigb[tid]       = sig_b[tid];
  s_sigb[tid + 128] = sig_b[tid + 128];
  __syncthreads();

  const bf16_t* encT = wts + OFF_ENCT;
  const bf16_t* mu1T = wts + OFF_MU1T;
  const bf16_t* mu2T = wts + OFF_MU2T;
  const bf16_t* decT = wts + OFF_DECT;
  const bf16_t* w2T  = wts + OFF_W2T;

  const int rowBase = blockIdx.x * ROWS_PER_BLOCK + wave * 16;
  bf16_t* zbf = s_zbf[wave];
  bf16_t* a2  = s_a2[wave];
  bf16_t* hb  = s_a2[wave];           // h (16x128) overlays a2 region (phases don't overlap)
  float*  dwb = s_dw[wave];

  const int nlocal = lane & 15;           // accumulator column within 16-wide tile
  const int hi8    = (lane & 16) ? 8 : 0; // accumulator row offset for hi half-wave

  // ---- load y tile (16x128 f32) as bf16 A-tile into LDS
  for (int idx = lane; idx < 16 * 128; idx += 32) {
    float v = y[(size_t)(rowBase + (idx >> 7)) * 128 + (idx & 127)];
    zbf[idx] = (bf16_t)v;
  }

  // ---- encoder: z0 = gelu(y @ enc_w + enc_b), kept in f32 accumulators
  v8f zacc[8];
#pragma unroll
  for (int t = 0; t < 8; ++t) {
    float bv = s_encb[t * 16 + nlocal];
#pragma unroll
    for (int i = 0; i < 8; ++i) zacc[t][i] = bv;
  }
  gemm_tiles<4>(zbf, 128, encT, 128, lane, zacc);
#pragma unroll
  for (int t = 0; t < 8; ++t) zacc[t] = gelu8(zacc[t]);

  // ---- 200 Euler-Maruyama steps
  for (int s = 0; s < NSTEPS; ++s) {
    // dW for this wave's 16 rows: one f32 per lane, broadcast via wave-private LDS
    {
      float xi = noise[((size_t)s * BATCH + rowBase + (lane >> 1)) * 2 + (lane & 1)];
      dwb[lane] = xi * SQRT_DT_F;
    }
    v8f dwv0, dwv1;
#pragma unroll
    for (int i = 0; i < 8; ++i) {
      int r = i + hi8;
      dwv0[i] = dwb[r * 2 + 0];
      dwv1[i] = dwb[r * 2 + 1];
    }

    // Phase A: stage old z as bf16 (for mu path) and A2 = z*dW pairs (for sigma path)
#pragma unroll
    for (int t = 0; t < 8; ++t) {
      int n = t * 16 + nlocal;
      v8f z  = zacc[t];
      v8f p0 = z * dwv0;   // v_pk_mul_f32 pairs
      v8f p1 = z * dwv1;
#pragma unroll
      for (int i = 0; i < 8; ++i) {
        int row = i + hi8;
        zbf[row * 128 + n] = (bf16_t)z[i];
        union { bf16_t b[2]; unsigned u; } pk;
        pk.b[0] = (bf16_t)p0[i];
        pk.b[1] = (bf16_t)p1[i];
        *(unsigned*)&a2[row * 256 + 2 * n] = pk.u;  // one packed ds_store_b32
      }
    }

    // Phase B: diffusion contribution — K=256 GEMM (A2 x W2), bias = sig_b . dW
    {
      v8f acc[8];
#pragma unroll
      for (int t = 0; t < 8; ++t) {
        int d = t * 16 + nlocal;
        acc[t] = s_sigb[2 * d] * dwv0 + s_sigb[2 * d + 1] * dwv1;
      }
      gemm_tiles<8>(a2, 256, w2T, 256, lane, acc);
#pragma unroll
      for (int t = 0; t < 8; ++t) zacc[t] += acc[t];
    }

    // Phase C: h = gelu(z_old @ mu_w1 + b1) -> bf16 into LDS (overlays a2; B is done)
    {
      v8f acc[8];
#pragma unroll
      for (int t = 0; t < 8; ++t) {
        float bv = s_b1[t * 16 + nlocal];
#pragma unroll
        for (int i = 0; i < 8; ++i) acc[t][i] = bv;
      }
      gemm_tiles<4>(zbf, 128, mu1T, 128, lane, acc);
#pragma unroll
      for (int t = 0; t < 8; ++t) {
        v8f g = gelu8(acc[t]);
        int n = t * 16 + nlocal;
#pragma unroll
        for (int i = 0; i < 8; ++i) hb[(i + hi8) * 128 + n] = (bf16_t)g[i];
      }
    }

    // Phase D: mu = h @ mu_w2 + b2 ; z += DT * mu
    {
      v8f acc[8];
#pragma unroll
      for (int t = 0; t < 8; ++t) {
        float bv = s_b2[t * 16 + nlocal];
#pragma unroll
        for (int i = 0; i < 8; ++i) acc[t][i] = bv;
      }
      gemm_tiles<4>(hb, 128, mu2T, 128, lane, acc);
#pragma unroll
      for (int t = 0; t < 8; ++t) zacc[t] += DT_F * acc[t];  // v_pk_fma_f32
    }
  }

  // ---- decoder: out = zT @ dec_w + dec_b
#pragma unroll
  for (int t = 0; t < 8; ++t) {
    int n = t * 16 + nlocal;
#pragma unroll
    for (int i = 0; i < 8; ++i) zbf[(i + hi8) * 128 + n] = (bf16_t)zacc[t][i];
  }
  {
    v8f acc[8];
#pragma unroll
    for (int t = 0; t < 8; ++t) {
      float bv = s_decb[t * 16 + nlocal];
#pragma unroll
      for (int i = 0; i < 8; ++i) acc[t][i] = bv;
    }
    gemm_tiles<4>(zbf, 128, decT, 128, lane, acc);
#pragma unroll
    for (int t = 0; t < 8; ++t) {
      int n = t * 16 + nlocal;
#pragma unroll
      for (int i = 0; i < 8; ++i)
        out[(size_t)(rowBase + i + hi8) * 128 + n] = acc[t][i];
    }
  }
}

extern "C" void kernel_launch(void* const* d_in, const int* in_sizes, int n_in,
                              void* d_out, int out_size, void* d_ws, size_t ws_size,
                              hipStream_t stream) {
  const float* y     = (const float*)d_in[0];
  const float* noise = (const float*)d_in[1];
  const float* enc_w = (const float*)d_in[2];
  const float* enc_b = (const float*)d_in[3];
  const float* mu_w1 = (const float*)d_in[4];
  const float* mu_b1 = (const float*)d_in[5];
  const float* mu_w2 = (const float*)d_in[6];
  const float* mu_b2 = (const float*)d_in[7];
  const float* sig_w = (const float*)d_in[8];
  const float* sig_b = (const float*)d_in[9];
  const float* dec_w = (const float*)d_in[10];
  const float* dec_b = (const float*)d_in[11];
  bf16_t* wts = (bf16_t*)d_ws;

  sde_prep_kernel<<<(WTS_ELEMS + 255) / 256, 256, 0, stream>>>(
      enc_w, mu_w1, mu_w2, dec_w, sig_w, wts);
  sde_kernel<<<BATCH / ROWS_PER_BLOCK, 128, 0, stream>>>(
      y, noise, enc_b, mu_b1, mu_b2, sig_b, dec_b, wts, (float*)d_out);
}
